// test_vgg_90915867722422
// MI455X (gfx1250) — matrix operand order repaired and checked by reading, not compiled, not run on hardware
//
#include <hip/hip_runtime.h>

// ---------------------------------------------------------------------------
// CDNA5 / gfx1250 VGG-16 forward: bf16 WMMA implicit-GEMM + TDM weight tiles.
// ---------------------------------------------------------------------------

typedef __attribute__((ext_vector_type(16))) __bf16 v16bf;
typedef __attribute__((ext_vector_type(8)))  __bf16 v8bf;
typedef __attribute__((ext_vector_type(8)))  float  v8f;
typedef __attribute__((ext_vector_type(4)))  unsigned int v4u;
typedef __attribute__((ext_vector_type(4)))  int    v4i;
typedef __attribute__((ext_vector_type(8)))  int    v8i;

#if defined(__has_builtin)
#if __has_builtin(__builtin_amdgcn_tensor_load_to_lds) && __has_builtin(__builtin_amdgcn_s_wait_tensorcnt)
#define USE_TDM 1
#endif
#endif

__device__ __forceinline__ v8f wmma_bf16(v16bf a, v16bf b, v8f c) {
  // 8 args: (neg_a, A, neg_b, B, c_mod, C, reuse_a, reuse_b)
  return __builtin_amdgcn_wmma_f32_16x16x32_bf16(false, a, false, b,
                                                 (short)0, c, false, false);
}

// ---------------- data-layout conversion kernels ----------------------------

// x: (32,3,224,224) f32 NCHW  ->  bf16 NHWC (32,224,224,3)
__global__ void nchw_to_nhwc(const float* __restrict__ src, __bf16* __restrict__ dst) {
  size_t i = (size_t)blockIdx.x * blockDim.x + threadIdx.x;
  const size_t total = (size_t)32 * 224 * 224 * 3;
  if (i >= total) return;
  int c = (int)(i % 3); size_t t = i / 3;
  int x = (int)(t % 224); t /= 224;
  int y = (int)(t % 224);
  int n = (int)(t / 224);
  dst[i] = (__bf16)src[(((size_t)n * 3 + c) * 224 + y) * 224 + x];
}

// conv weight (oc,ic,3,3) f32 -> bf16 [K = (ky*3+kx)*ic + c][oc]
__global__ void conv_w_bf16(const float* __restrict__ src, __bf16* __restrict__ dst,
                            int ic, int oc) {
  int i = blockIdx.x * blockDim.x + threadIdx.x;
  int total = ic * 9 * oc;
  if (i >= total) return;
  int o = i % oc;
  int k = i / oc;
  int c  = k % ic;
  int kk = k / ic;                  // ky*3+kx
  dst[i] = (__bf16)src[((size_t)o * ic + c) * 9 + kk];
}

__global__ void zero_stats(float* __restrict__ a, float* __restrict__ b, int n) {
  int i = blockIdx.x * blockDim.x + threadIdx.x;
  if (i < n) { a[i] = 0.f; b[i] = 0.f; }
}

// ---------------- implicit-GEMM conv (128x64 tile, K-step 32) ---------------

#define CTM 128
#define CTN 64
#define CTK 32

__global__ __launch_bounds__(256)
void conv_wmma(const __bf16* __restrict__ in,   // NHWC bf16 (32,H,W,Cin)
               const __bf16* __restrict__ wt,   // [9*Cin][Cout] bf16
               __bf16* __restrict__ out,        // raw conv NHWC bf16 (32,H,W,Cout)
               float* __restrict__ sum, float* __restrict__ sumsq,
               int H, int W, int Cin, int Cout) {
  __shared__ __attribute__((aligned(16))) __bf16 As[CTM][CTK + 16]; // 96B rows
  __shared__ __attribute__((aligned(16))) __bf16 Bs[CTK][CTN];      // packed (TDM layout)

  const int Ktot = 9 * Cin;
  const int HW = H * W;
  const int tid  = threadIdx.x;
  const int m0   = blockIdx.x * CTM;        // pixel-tile base (always in range)
  const int n0   = blockIdx.y * CTN;        // out-channel tile base
  const int wv   = tid >> 5;
  const int lane = tid & 31;
  const int wm   = (wv & 3) * 32;           // wave M offset inside tile
  const int wn   = (wv >> 2) * 32;          // wave N offset inside tile
  const int lm   = lane & 15;
  const int half = lane >> 4;

  // ---- per-thread A-row coordinates, hoisted out of the K loop ----
  const bool fast = (Cin & 31) == 0;        // all layers except layer 0
  const int ar = tid >> 1;                  // A row handled by this thread
  const int ah = (tid & 1) * 16;            // 16-element half-chunk offset
  const int ap = m0 + ar;
  const int an = ap / HW;
  const int arem = ap - an * HW;
  const int ay = arem / W;
  const int ax = arem - ay * W;
  const __bf16* arow = in + (((size_t)an * H + ay) * W + ax) * Cin;
  int kidx = 0;                             // (ky*3+kx) slice of this half-chunk
  int cbase = ah;                           // channel base within slice

  v8f acc[2][2];
  const v8f vzero = {0.f, 0.f, 0.f, 0.f, 0.f, 0.f, 0.f, 0.f};
#pragma unroll
  for (int i = 0; i < 2; ++i)
#pragma unroll
    for (int j = 0; j < 2; ++j) acc[i][j] = vzero;

  const v8bf bzero = {(__bf16)0.f, (__bf16)0.f, (__bf16)0.f, (__bf16)0.f,
                      (__bf16)0.f, (__bf16)0.f, (__bf16)0.f, (__bf16)0.f};

  for (int k0 = 0; k0 < Ktot; k0 += CTK) {
#ifdef USE_TDM
    // ---- Tensor Data Mover: B tile (32 x 64 bf16) global -> LDS, wave 0 ----
    if (wv == 0) {
      unsigned long long ga =
          (unsigned long long)(uintptr_t)(wt + ((size_t)k0 * Cout + n0));
      unsigned ldsa = (unsigned)(uintptr_t)(&Bs[0][0]);
      int dim0 = Cout - n0;                 // remaining columns
      int dim1 = Ktot - k0;                 // remaining K rows (zero-fill past end)
      v4u g0;
      g0[0] = 1u;                                        // count=1, user mode
      g0[1] = ldsa;                                      // lds_addr
      g0[2] = (unsigned)ga;                              // global_addr[31:0]
      g0[3] = (unsigned)(ga >> 32) | (2u << 30);         // addr[56:32] | type=2
      v8i g1;
      g1[0] = 0x10000;                                   // data_size = 2 bytes
      g1[1] = (dim0 & 0xffff) << 16;                     // tensor_dim0[15:0]
      g1[2] = ((dim0 >> 16) & 0xffff) | ((dim1 & 0xffff) << 16);
      g1[3] = ((dim1 >> 16) & 0xffff) | (64 << 16);      // tile_dim0 = 64
      g1[4] = 32;                                        // tile_dim1 = 32
      g1[5] = Cout;                                      // dim0_stride[31:0]
      g1[6] = 0;
      g1[7] = 0;
      v4i g2 = {0, 0, 0, 0}, g3 = {0, 0, 0, 0};
#if __clang_major__ >= 23
      v8i g4 = {0, 0, 0, 0, 0, 0, 0, 0};
      __builtin_amdgcn_tensor_load_to_lds(g0, g1, g2, g3, g4, 0);
#else
      __builtin_amdgcn_tensor_load_to_lds(g0, g1, g2, g3, 0);
#endif
    }
#endif
    if (tid == 0 && k0 + CTK < Ktot)
      __builtin_prefetch(wt + (size_t)(k0 + CTK) * Cout + n0, 0, 2); // global_prefetch_b8

    // ---- im2col A tile: 128x32 bf16 ----
    if (fast) {
      // 16 consecutive k == 16 consecutive channels: two b128 loads per thread
      int ky = kidx / 3 - 1, kx = kidx - (kidx / 3) * 3 - 1;
      int iy = ay + ky, ix = ax + kx;
      v8bf lo = bzero, hi = bzero;
      if ((unsigned)iy < (unsigned)H && (unsigned)ix < (unsigned)W) {
        const v8bf* sp = (const v8bf*)(arow + (ky * W + kx) * Cin + cbase);
        lo = sp[0];
        hi = sp[1];
      }
      *(v8bf*)&As[ar][ah]     = lo;
      *(v8bf*)&As[ar][ah + 8] = hi;
      cbase += CTK;
      if (cbase >= Cin) { cbase -= Cin; ++kidx; }
    } else {
      // generic path (layer 0: Cin=3, Ktot=27, single chunk)
#pragma unroll
      for (int i = 0; i < 16; ++i) {
        int idx = tid + i * 256;
        int r  = idx >> 5;
        int kk = idx & 31;
        int k  = k0 + kk;
        __bf16 v = (__bf16)0.f;
        if (k < Ktot) {
          int c    = k % Cin;
          int kidg = k / Cin;
          int ky = kidg / 3 - 1, kx = kidg % 3 - 1;
          int p   = m0 + r;
          int n   = p / HW;
          int rem = p - n * HW;
          int y   = rem / W;
          int x   = rem - y * W;
          int iy = y + ky, ix = x + kx;
          if ((unsigned)iy < (unsigned)H && (unsigned)ix < (unsigned)W)
            v = in[(((size_t)n * H + iy) * W + ix) * Cin + c];
        }
        As[r][kk] = v;
      }
    }

#ifndef USE_TDM
    // ---- fallback B tile load: one b128 per thread ----
    {
      int kk = tid >> 3;                    // 0..31
      int nn = (tid & 7) * 8;
      v8bf v = bzero;
      if (k0 + kk < Ktot)
        v = *(const v8bf*)(wt + (size_t)(k0 + kk) * Cout + n0 + nn);
      *(v8bf*)&Bs[kk][nn] = v;
    }
#endif

#ifdef USE_TDM
    if (wv == 0) __builtin_amdgcn_s_wait_tensorcnt(0);
#endif
    __syncthreads();

    // ---- fragments (ISA 16-bit A/B layouts); A via 2x ds_load_b128 ----
    v16bf a[2], b[2];
#pragma unroll
    for (int t = 0; t < 2; ++t) {
      int m = wm + t * 16 + lm;
      v8bf alo = *(const v8bf*)&As[m][half * 8];
      v8bf ahi = *(const v8bf*)&As[m][16 + half * 8];
      a[t] = __builtin_shufflevector(alo, ahi, 0, 1, 2, 3, 4, 5, 6, 7,
                                     8, 9, 10, 11, 12, 13, 14, 15);
      int n = wn + t * 16 + lm;
#pragma unroll
      for (int v = 0; v < 8; ++v) {
        int kr = (v >> 2) * 16 + half * 8 + 2 * (v & 3);
        b[t][2 * v]     = Bs[kr][n];
        b[t][2 * v + 1] = Bs[kr + 1][n];
      }
    }
#pragma unroll
    for (int i = 0; i < 2; ++i)
#pragma unroll
      for (int j = 0; j < 2; ++j)
        acc[i][j] = wmma_bf16(a[i], b[j], acc[i][j]);
    __syncthreads();
  }

  // ---- epilogue: raw conv bf16 + BN partial sums ----
#pragma unroll
  for (int i = 0; i < 2; ++i) {
#pragma unroll
    for (int j = 0; j < 2; ++j) {
      int ch = n0 + wn + j * 16 + lm;
      float s = 0.f, ss = 0.f;
#pragma unroll
      for (int r = 0; r < 8; ++r) {
        float v = acc[i][j][r];
        int m = m0 + wm + i * 16 + half * 8 + r;
        out[(size_t)m * Cout + ch] = (__bf16)v;
        s += v; ss += v * v;
      }
      atomicAdd(&sum[ch], s);
      atomicAdd(&sumsq[ch], ss);
    }
  }
}

// ---------------- batch-norm helpers ---------------------------------------

__global__ void bn_finalize(const float* __restrict__ sum, const float* __restrict__ sumsq,
                            const float* __restrict__ gamma, const float* __restrict__ beta,
                            float* __restrict__ scale, float* __restrict__ shift,
                            int C, float inv_cnt) {
  int c = blockIdx.x * blockDim.x + threadIdx.x;
  if (c >= C) return;
  float mean = sum[c] * inv_cnt;
  float var  = sumsq[c] * inv_cnt - mean * mean;
  float sc   = gamma[c] * rsqrtf(var + 1e-5f);
  scale[c] = sc;
  shift[c] = beta[c] - mean * sc;
}

// processes 8 bf16 per thread (total8 = total/8; C % 8 == 0)
__global__ void bn_relu_inplace(__bf16* __restrict__ buf,
                                const float* __restrict__ scale,
                                const float* __restrict__ shift,
                                int C, size_t total8) {
  size_t i = (size_t)blockIdx.x * blockDim.x + threadIdx.x;
  if (i >= total8) return;
  size_t e = i * 8;
  int c0 = (int)(e % C);
  v8bf v = *(v8bf*)(buf + e);
  v8bf r;
#pragma unroll
  for (int j = 0; j < 8; ++j) {
    float f = (float)v[j] * scale[c0 + j] + shift[c0 + j];
    r[j] = (__bf16)fmaxf(f, 0.f);
  }
  *(v8bf*)(buf + e) = r;
}

// relu(maxpool(bn(x))) == maxpool(relu(bn(x))) since relu is monotone
__global__ void bn_relu_pool(const __bf16* __restrict__ src, __bf16* __restrict__ dst,
                             const float* __restrict__ scale, const float* __restrict__ shift,
                             int H, int W, int C, size_t total) {
  size_t i = (size_t)blockIdx.x * blockDim.x + threadIdx.x;
  if (i >= total) return;
  int c = (int)(i % C); size_t t = i / C;
  int W2 = W >> 1, H2 = H >> 1;
  int x2 = (int)(t % W2); t /= W2;
  int y2 = (int)(t % H2);
  int n  = (int)(t / H2);
  float sc = scale[c], sh = shift[c];
  float m = -3.4e38f;
#pragma unroll
  for (int dy = 0; dy < 2; ++dy)
#pragma unroll
    for (int dx = 0; dx < 2; ++dx) {
      float v = (float)src[(((size_t)n * H + (2 * y2 + dy)) * W + (2 * x2 + dx)) * C + c];
      v = v * sc + sh;
      m = fmaxf(m, v);
    }
  dst[i] = (__bf16)fmaxf(m, 0.f);
}

__global__ void avgpool2x2(const __bf16* __restrict__ src, __bf16* __restrict__ dst,
                           int H, int W, int C, size_t total) {
  size_t i = (size_t)blockIdx.x * blockDim.x + threadIdx.x;
  if (i >= total) return;
  int c = (int)(i % C); size_t t = i / C;
  int W2 = W >> 1, H2 = H >> 1;
  int x2 = (int)(t % W2); t /= W2;
  int y2 = (int)(t % H2);
  int n  = (int)(t / H2);
  float s = 0.f;
#pragma unroll
  for (int dy = 0; dy < 2; ++dy)
#pragma unroll
    for (int dx = 0; dx < 2; ++dx)
      s += (float)src[(((size_t)n * H + (2 * y2 + dy)) * W + (2 * x2 + dx)) * C + c];
  dst[i] = (__bf16)(s * 0.25f);
}

// ---------------- FC layers (M=32 WMMA GEMMs) --------------------------------

// act: bf16 (32,7,7,512) NHWC ; w: f32 (25088, 512) with k = c*49 + (h*7+w)
__global__ __launch_bounds__(128)
void fc1_wmma(const __bf16* __restrict__ act, const float* __restrict__ w,
              float* __restrict__ out /* 32x512 f32 */) {
  const int tid = threadIdx.x, wv = tid >> 5, lane = tid & 31;
  const int n0 = blockIdx.x * 64 + wv * 16;
  const int lm = lane & 15, half = lane >> 4;
  const v8f vz = {0.f,0.f,0.f,0.f,0.f,0.f,0.f,0.f};
  v8f acc0 = vz, acc1 = vz;
  for (int k0 = 0; k0 < 25088; k0 += 32) {
    v16bf a0, a1, b;
#pragma unroll
    for (int v = 0; v < 8; ++v) {
      int kb = (v < 4) ? (half * 8 + 2 * v) : (16 + half * 8 + 2 * (v - 4));
      int k = k0 + kb;
      int c = k / 49,  p  = k - c * 49;      // torch .view flatten: k = c*49 + p
      int k1 = k + 1;
      int c1 = k1 / 49, p1 = k1 - c1 * 49;
      a0[2*v]   = act[((size_t)lm        * 49 + p ) * 512 + c ];
      a0[2*v+1] = act[((size_t)lm        * 49 + p1) * 512 + c1];
      a1[2*v]   = act[((size_t)(lm + 16) * 49 + p ) * 512 + c ];
      a1[2*v+1] = act[((size_t)(lm + 16) * 49 + p1) * 512 + c1];
      int kr = k0 + kb;
      b[2*v]   = (__bf16)w[(size_t)kr       * 512 + n0 + lm];
      b[2*v+1] = (__bf16)w[(size_t)(kr + 1) * 512 + n0 + lm];
    }
    acc0 = wmma_bf16(a0, b, acc0);
    acc1 = wmma_bf16(a1, b, acc1);
  }
#pragma unroll
  for (int r = 0; r < 8; ++r) {
    out[(size_t)(half * 8 + r)      * 512 + n0 + lm] = acc0[r];
    out[(size_t)(16 + half * 8 + r) * 512 + n0 + lm] = acc1[r];
  }
}

__global__ void bn1d_relu(float* __restrict__ x /*32x512*/,
                          const float* __restrict__ gamma,
                          const float* __restrict__ beta) {
  int c = blockIdx.x * blockDim.x + threadIdx.x;
  if (c >= 512) return;
  float s = 0.f, ss = 0.f;
  for (int n = 0; n < 32; ++n) { float v = x[n * 512 + c]; s += v; ss += v * v; }
  float mean = s * (1.f / 32.f);
  float var  = ss * (1.f / 32.f) - mean * mean;
  float sc = gamma[c] * rsqrtf(var + 1e-5f);
  float sh = beta[c] - mean * sc;
  for (int n = 0; n < 32; ++n)
    x[n * 512 + c] = fmaxf(x[n * 512 + c] * sc + sh, 0.f);
}

// act: f32 (32,512) ; w: f32 (512,1000) ; out: f32 (32,1000)
__global__ __launch_bounds__(128)
void fc2_wmma(const float* __restrict__ act, const float* __restrict__ w,
              float* __restrict__ out) {
  const int tid = threadIdx.x, wv = tid >> 5, lane = tid & 31;
  const int n0 = blockIdx.x * 64 + wv * 16;
  const int lm = lane & 15, half = lane >> 4;
  const int n = n0 + lm;
  const bool ok = (n < 1000);
  const v8f vz = {0.f,0.f,0.f,0.f,0.f,0.f,0.f,0.f};
  v8f acc0 = vz, acc1 = vz;
  for (int k0 = 0; k0 < 512; k0 += 32) {
    v16bf a0, a1, b;
#pragma unroll
    for (int v = 0; v < 8; ++v) {
      int kb = (v < 4) ? (half * 8 + 2 * v) : (16 + half * 8 + 2 * (v - 4));
      int k = k0 + kb;
      a0[2*v]   = (__bf16)act[(size_t)lm        * 512 + k];
      a0[2*v+1] = (__bf16)act[(size_t)lm        * 512 + k + 1];
      a1[2*v]   = (__bf16)act[(size_t)(lm + 16) * 512 + k];
      a1[2*v+1] = (__bf16)act[(size_t)(lm + 16) * 512 + k + 1];
      int kr = k0 + kb;
      b[2*v]   = ok ? (__bf16)w[(size_t)kr       * 1000 + n] : (__bf16)0.f;
      b[2*v+1] = ok ? (__bf16)w[(size_t)(kr + 1) * 1000 + n] : (__bf16)0.f;
    }
    acc0 = wmma_bf16(a0, b, acc0);
    acc1 = wmma_bf16(a1, b, acc1);
  }
  if (ok) {
#pragma unroll
    for (int r = 0; r < 8; ++r) {
      out[(size_t)(half * 8 + r)      * 1000 + n] = acc0[r];
      out[(size_t)(16 + half * 8 + r) * 1000 + n] = acc1[r];
    }
  }
}

// ---------------- host orchestration ----------------------------------------

extern "C" void kernel_launch(void* const* d_in, const int* in_sizes, int n_in,
                              void* d_out, int out_size, void* d_ws, size_t ws_size,
                              hipStream_t stream) {
  (void)in_sizes; (void)n_in; (void)out_size; (void)ws_size;
  const float* x = (const float*)d_in[0];
  const float* cw[13]; const float* gm[13]; const float* bt[13];
  for (int i = 0; i < 13; ++i) {
    cw[i] = (const float*)d_in[1 + i];
    gm[i] = (const float*)d_in[14 + i];
    bt[i] = (const float*)d_in[27 + i];
  }
  const float* fc1w = (const float*)d_in[40];
  const float* g1d  = (const float*)d_in[41];
  const float* b1d  = (const float*)d_in[42];
  const float* fc2w = (const float*)d_in[43];
  float* out = (float*)d_out;

  char* base = (char*)d_ws; size_t off = 0;
  auto carve = [&](size_t b) -> void* {
    void* p = base + off; off = (off + b + 255) & ~(size_t)255; return p;
  };
  const size_t ACT = (size_t)32 * 224 * 224 * 64 * 2;  // largest activation (bf16)
  __bf16* bufA  = (__bf16*)carve(ACT);
  __bf16* bufB  = (__bf16*)carve(ACT);
  __bf16* wbuf  = (__bf16*)carve((size_t)14710464 * 2);  // all conv weights, bf16
  float*  sum   = (float*)carve(512 * 4);
  float*  sumsq = (float*)carve(512 * 4);
  float*  scale = (float*)carve(512 * 4);
  float*  shift = (float*)carve(512 * 4);
  float*  fc1o  = (float*)carve(32 * 512 * 4);

  static const int  CH[13][2] = {{3,64},{64,64},{64,128},{128,128},{128,256},{256,256},
                                 {256,256},{256,512},{512,512},{512,512},{512,512},
                                 {512,512},{512,512}};
  static const bool POOL[13]  = {false,true,false,true,false,false,true,
                                 false,false,true,false,false,false};

  // layout conversions (re-run every call: deterministic)
  size_t woff = 0;
  for (int i = 0; i < 13; ++i) {
    int ic = CH[i][0], oc = CH[i][1], tot = ic * 9 * oc;
    conv_w_bf16<<<(tot + 255) / 256, 256, 0, stream>>>(cw[i], wbuf + woff, ic, oc);
    woff += (size_t)tot;
  }
  {
    size_t tot = (size_t)32 * 224 * 224 * 3;
    nchw_to_nhwc<<<(unsigned)((tot + 255) / 256), 256, 0, stream>>>(x, bufA);
  }

  __bf16* cur = bufA;
  __bf16* oth = bufB;
  int H = 224, W = 224;
  woff = 0;
  for (int i = 0; i < 13; ++i) {
    int ic = CH[i][0], oc = CH[i][1];
    size_t M = (size_t)32 * H * W;                    // always a multiple of 128
    zero_stats<<<2, 256, 0, stream>>>(sum, sumsq, 512);
    dim3 grid((unsigned)(M / 128), (unsigned)(oc / 64));
    conv_wmma<<<grid, 256, 0, stream>>>(cur, wbuf + woff, oth, sum, sumsq, H, W, ic, oc);
    bn_finalize<<<2, 256, 0, stream>>>(sum, sumsq, gm[i], bt[i], scale, shift, oc,
                                       (float)(1.0 / (double)M));
    if (POOL[i]) {
      size_t tot = (size_t)32 * (H / 2) * (W / 2) * oc;
      bn_relu_pool<<<(unsigned)((tot + 255) / 256), 256, 0, stream>>>(
          oth, cur, scale, shift, H, W, oc, tot);     // pooled -> cur
      H >>= 1; W >>= 1;
    } else {
      size_t tot8 = (M * (size_t)oc) / 8;
      bn_relu_inplace<<<(unsigned)((tot8 + 255) / 256), 256, 0, stream>>>(
          oth, scale, shift, oc, tot8);
      __bf16* t = cur; cur = oth; oth = t;
    }
    woff += (size_t)ic * 9 * oc;
  }

  // avgpool 14x14 -> 7x7 (cur -> oth)
  {
    size_t tot = (size_t)32 * 7 * 7 * 512;
    avgpool2x2<<<(unsigned)((tot + 255) / 256), 256, 0, stream>>>(cur, oth, 14, 14, 512, tot);
  }
  fc1_wmma<<<8, 128, 0, stream>>>(oth, fc1w, fc1o);
  bn1d_relu<<<2, 256, 0, stream>>>(fc1o, g1d, b1d);
  fc2_wmma<<<16, 128, 0, stream>>>(fc1o, fc2w, out);
}
